// SRGNNCell_30751965840099
// MI455X (gfx1250) — compile-verified, matrix operand order checked by reading, not run on hardware
//
#include <hip/hip_runtime.h>
#include <math.h>

// ---------------------------------------------------------------------------
// SR-GNN cell for MI455X (gfx1250, wave32, WMMA).
// bf16 WMMA (f32 accumulate) for all GEMMs; f32 atomics for the edge scatter.
// ---------------------------------------------------------------------------

typedef __attribute__((ext_vector_type(16))) __bf16 v16bf;
typedef __attribute__((ext_vector_type(8)))  float  v8f;

#define DIM 64

static __device__ __forceinline__ v8f wmma_bf16(v16bf a, v16bf b, v8f c) {
  // D = A(16x32 bf16) * B(32x16 bf16) + C(16x16 f32)
  return __builtin_amdgcn_wmma_f32_16x16x32_bf16(
      /*neg_a=*/false, a, /*neg_b=*/false, b,
      /*c_mod=*/(short)0, c, /*reuse_a=*/false, /*reuse_b=*/false);
}

// Build a 16x32 (MxK) bf16 A-tile from a row-major f32 matrix with leading dim ld.
// Per ISA layout: lane l -> row m = l&15; khalf = l>>4;
//   elems 0..7  = K = k0 + khalf*8 + j
//   elems 8..15 = K = k0 + 16 + khalf*8 + j
// All offsets are multiples of 8 floats -> 32B-aligned float4 loads.
static __device__ __forceinline__ v16bf load_A(const float* __restrict__ src,
                                               int ld, int row0, int k0, int lane) {
  const int m  = lane & 15;
  const int kh = lane >> 4;
  const float4* q = (const float4*)(src + (size_t)(row0 + m) * ld + k0 + kh * 8);
  float4 x0 = q[0], x1 = q[1];  // K = k0+kh*8 .. +7
  float4 y0 = q[4], y1 = q[5];  // K = k0+16+kh*8 .. +7
  v16bf a;
  a[0]  = (__bf16)x0.x; a[1]  = (__bf16)x0.y; a[2]  = (__bf16)x0.z; a[3]  = (__bf16)x0.w;
  a[4]  = (__bf16)x1.x; a[5]  = (__bf16)x1.y; a[6]  = (__bf16)x1.z; a[7]  = (__bf16)x1.w;
  a[8]  = (__bf16)y0.x; a[9]  = (__bf16)y0.y; a[10] = (__bf16)y0.z; a[11] = (__bf16)y0.w;
  a[12] = (__bf16)y1.x; a[13] = (__bf16)y1.y; a[14] = (__bf16)y1.z; a[15] = (__bf16)y1.w;
  return a;
}

// B operand: pre-packed per-lane layout, one contiguous 32B load per lane.
static __device__ __forceinline__ v16bf load_B(const __bf16* __restrict__ pw,
                                               int block, int lane) {
  return *(const v16bf*)(pw + (size_t)block * 512 + lane * 16);
}

// Store a 16x16 f32 C/D tile: lane l -> col n = l&15, rows (l>>4)*8 + v.
static __device__ __forceinline__ void store_C(float* __restrict__ dst, int ld,
                                               int row0, int col0, v8f acc, int lane) {
  const int n  = lane & 15;
  const int mh = lane >> 4;
#pragma unroll
  for (int v = 0; v < 8; ++v)
    dst[(size_t)(row0 + mh * 8 + v) * ld + col0 + n] = acc[v];
}

static __device__ __forceinline__ float sigmoidf_(float x) {
  return 1.0f / (1.0f + __expf(-x));
}

// ---------------------------------------------------------------------------
// Kernel 1: repack a f32 weight matrix [K x Ncols] (row-major, K = 32*KC,
// Ncols = 16*NT) into bf16 WMMA-B per-lane layout. One 32-lane block per
// (kc, nt) tile; block id = kc*NT + nt.
// ---------------------------------------------------------------------------
__global__ void repack_kernel(const float* __restrict__ W, int Ncols,
                              __bf16* __restrict__ out) {
  const int blk  = blockIdx.x;
  const int NT   = Ncols >> 4;
  const int kc   = blk / NT;
  const int nt   = blk % NT;
  const int lane = threadIdx.x;
  const int n    = lane & 15;
  const int kh   = lane >> 4;
  v16bf t;
#pragma unroll
  for (int j = 0; j < 8; ++j)
    t[j] = (__bf16)W[(size_t)(kc * 32 + kh * 8 + j) * Ncols + nt * 16 + n];
#pragma unroll
  for (int j = 0; j < 8; ++j)
    t[8 + j] = (__bf16)W[(size_t)(kc * 32 + 16 + kh * 8 + j) * Ncols + nt * 16 + n];
  *(v16bf*)(out + (size_t)blk * 512 + lane * 16) = t;
}

// ---------------------------------------------------------------------------
// Kernel 2: zero the scatter accumulators.
// ---------------------------------------------------------------------------
__global__ void zero_kernel(float4* __restrict__ p, long n4) {
  long i = (long)blockIdx.x * blockDim.x + threadIdx.x;
  if (i < n4) p[i] = make_float4(0.f, 0.f, 0.f, 0.f);
}

// ---------------------------------------------------------------------------
// Kernel 3: t_in = hidden@W_in + b_in ; t_out = hidden@W_out + b_out.
// One wave per 16-row tile; A (hidden) shared between both weight sets.
// ---------------------------------------------------------------------------
__global__ __launch_bounds__(256) void transform_kernel(
    const float* __restrict__ hidden,
    const __bf16* __restrict__ pwA, const float* __restrict__ bA,
    const __bf16* __restrict__ pwB, const float* __restrict__ bB,
    float* __restrict__ tA, float* __restrict__ tB, int N) {
  const int lane = threadIdx.x & 31;
  const int wave = blockIdx.x * (blockDim.x >> 5) + (threadIdx.x >> 5);
  const int row0 = wave * 16;
  if (row0 + 16 > N) return;           // wave-uniform guard (N % 16 == 0 here)
  const int n = lane & 15;

  v16bf a0 = load_A(hidden, DIM, row0, 0, lane);
  v16bf a1 = load_A(hidden, DIM, row0, 32, lane);

#pragma unroll
  for (int nt = 0; nt < 4; ++nt) {
    float bv = bA[nt * 16 + n];
    v8f acc;
#pragma unroll
    for (int v = 0; v < 8; ++v) acc[v] = bv;
    acc = wmma_bf16(a0, load_B(pwA, 0 * 4 + nt, lane), acc);
    acc = wmma_bf16(a1, load_B(pwA, 1 * 4 + nt, lane), acc);
    store_C(tA, DIM, row0, nt * 16, acc, lane);

    bv = bB[nt * 16 + n];
    v8f acc2;
#pragma unroll
    for (int v = 0; v < 8; ++v) acc2[v] = bv;
    acc2 = wmma_bf16(a0, load_B(pwB, 0 * 4 + nt, lane), acc2);
    acc2 = wmma_bf16(a1, load_B(pwB, 1 * 4 + nt, lane), acc2);
    store_C(tB, DIM, row0, nt * 16, acc2, lane);
  }
}

// ---------------------------------------------------------------------------
// Kernel 4: edge-weighted scatter-sum: agg[dst] += t[src] * w.
// 16 lanes per edge, float4 gather, 4 non-returning f32 atomics per lane.
// ---------------------------------------------------------------------------
__global__ __launch_bounds__(256) void scatter_kernel(
    const float* __restrict__ t, const int* __restrict__ src,
    const int* __restrict__ dst, const float* __restrict__ w,
    float* __restrict__ agg, int E) {
  long gid = (long)blockIdx.x * blockDim.x + threadIdx.x;
  int e = (int)(gid >> 4);
  if (e >= E) return;
  int part = (int)(gid & 15);
  int s = src[e];
  int d = dst[e];
  float wt = w[e];
  float4 v = ((const float4*)(t + (size_t)s * DIM))[part];
  float* out = agg + (size_t)d * DIM + part * 4;
  atomicAdd(out + 0, v.x * wt);
  atomicAdd(out + 1, v.y * wt);
  atomicAdd(out + 2, v.z * wt);
  atomicAdd(out + 3, v.w * wt);
}

// ---------------------------------------------------------------------------
// Kernel 5: fused GRU cell. Per 16-row tile:
//   gi = [agg_in|agg_out] @ W_ih + b_ih   (K=128, 192 cols)
//   gh = hidden @ W_hh + b_hh             (K=64,  192 cols)
// r/i gate columns (0..127) of gi and gh share accumulators; i_n and h_n
// (cols 128..191) kept separate. Gating done in registers; hy written out.
// ---------------------------------------------------------------------------
__global__ __launch_bounds__(256) void gru_kernel(
    const float* __restrict__ hidden,
    const float* __restrict__ agg_in, const float* __restrict__ agg_out,
    const __bf16* __restrict__ pw_ih, const float* __restrict__ b_ih,
    const __bf16* __restrict__ pw_hh, const float* __restrict__ b_hh,
    float* __restrict__ hy, int N) {
  const int lane = threadIdx.x & 31;
  const int wave = blockIdx.x * (blockDim.x >> 5) + (threadIdx.x >> 5);
  const int row0 = wave * 16;
  if (row0 + 16 > N) return;
  const int n  = lane & 15;
  const int mh = lane >> 4;

  // A operands
  v16bf ai0 = load_A(agg_in,  DIM, row0, 0,  lane);
  v16bf ai1 = load_A(agg_in,  DIM, row0, 32, lane);
  v16bf ao0 = load_A(agg_out, DIM, row0, 0,  lane);
  v16bf ao1 = load_A(agg_out, DIM, row0, 32, lane);
  v16bf ah0 = load_A(hidden,  DIM, row0, 0,  lane);
  v16bf ah1 = load_A(hidden,  DIM, row0, 32, lane);

  // acc[0..7]: (gi+gh) cols 0..127 ; acc[8..11]: i_n ; hn[0..3]: h_n
  v8f acc[12], hn[4];
#pragma unroll
  for (int nt = 0; nt < 12; ++nt) {
    float bv = b_ih[nt * 16 + n] + ((nt < 8) ? b_hh[nt * 16 + n] : 0.0f);
#pragma unroll
    for (int v = 0; v < 8; ++v) acc[nt][v] = bv;
  }
#pragma unroll
  for (int tt = 0; tt < 4; ++tt) {
    float bv = b_hh[(8 + tt) * 16 + n];
#pragma unroll
    for (int v = 0; v < 8; ++v) hn[tt][v] = bv;
  }

  // gi: K-chunks = {agg_in[0:32], agg_in[32:64], agg_out[0:32], agg_out[32:64]}
#pragma unroll
  for (int kc = 0; kc < 4; ++kc) {
    v16bf a = (kc == 0) ? ai0 : (kc == 1) ? ai1 : (kc == 2) ? ao0 : ao1;
#pragma unroll
    for (int nt = 0; nt < 12; ++nt)
      acc[nt] = wmma_bf16(a, load_B(pw_ih, kc * 12 + nt, lane), acc[nt]);
  }
  // gh: K-chunks over hidden
#pragma unroll
  for (int kc = 0; kc < 2; ++kc) {
    v16bf a = (kc == 0) ? ah0 : ah1;
#pragma unroll
    for (int nt = 0; nt < 12; ++nt) {
      if (nt < 8)
        acc[nt] = wmma_bf16(a, load_B(pw_hh, kc * 12 + nt, lane), acc[nt]);
      else
        hn[nt - 8] = wmma_bf16(a, load_B(pw_hh, kc * 12 + nt, lane), hn[nt - 8]);
    }
  }

  // Gating + output. Tile column c = t*16 + n, row r = row0 + mh*8 + v.
#pragma unroll
  for (int v = 0; v < 8; ++v) {
    const size_t r = (size_t)(row0 + mh * 8 + v);
#pragma unroll
    for (int tt = 0; tt < 4; ++tt) {
      float rg = sigmoidf_(acc[tt][v]);            // reset gate (cols 0..63)
      float ig = sigmoidf_(acc[4 + tt][v]);        // input gate (cols 64..127)
      float ng = tanhf(acc[8 + tt][v] + rg * hn[tt][v]);
      float h  = hidden[r * DIM + tt * 16 + n];
      hy[r * DIM + tt * 16 + n] = (1.0f - ig) * h + ig * ng;
    }
  }
}

// ---------------------------------------------------------------------------
// Host-side launcher.
// ---------------------------------------------------------------------------
extern "C" void kernel_launch(void* const* d_in, const int* in_sizes, int n_in,
                              void* d_out, int out_size, void* d_ws, size_t ws_size,
                              hipStream_t stream) {
  const float* hidden  = (const float*)d_in[0];
  const int*   in_src  = (const int*)  d_in[1];
  const int*   in_dst  = (const int*)  d_in[2];
  const float* in_w    = (const float*)d_in[3];
  const int*   out_src = (const int*)  d_in[4];
  const int*   out_dst = (const int*)  d_in[5];
  const float* out_w   = (const float*)d_in[6];
  const float* W_in    = (const float*)d_in[7];
  const float* b_in    = (const float*)d_in[8];
  const float* W_out   = (const float*)d_in[9];
  const float* b_out   = (const float*)d_in[10];
  const float* W_ih    = (const float*)d_in[11];
  const float* b_ih    = (const float*)d_in[12];
  const float* W_hh    = (const float*)d_in[13];
  const float* b_hh    = (const float*)d_in[14];
  float* hy = (float*)d_out;

  const int N = in_sizes[0] / DIM;
  const int E = in_sizes[1];

  // Workspace carve-up (all offsets keep >=32B alignment).
  char* ws = (char*)d_ws;
  size_t off = 0;
  float* t_in    = (float*)(ws + off); off += (size_t)N * DIM * sizeof(float);
  float* t_out   = (float*)(ws + off); off += (size_t)N * DIM * sizeof(float);
  float* agg_in  = (float*)(ws + off); off += (size_t)N * DIM * sizeof(float);
  float* agg_out = (float*)(ws + off); off += (size_t)N * DIM * sizeof(float);
  __bf16* pw_in  = (__bf16*)(ws + off); off += 8  * 1024;  // 64x64 : 2x4 tiles
  __bf16* pw_out = (__bf16*)(ws + off); off += 8  * 1024;
  __bf16* pw_ih  = (__bf16*)(ws + off); off += 48 * 1024;  // 128x192: 4x12 tiles
  __bf16* pw_hh  = (__bf16*)(ws + off); off += 24 * 1024;  // 64x192 : 2x12 tiles

  // 1) Repack weights into bf16 WMMA-B layout (tiny, L2-resident afterwards).
  repack_kernel<<<8,  32, 0, stream>>>(W_in,  64,  pw_in);
  repack_kernel<<<8,  32, 0, stream>>>(W_out, 64,  pw_out);
  repack_kernel<<<48, 32, 0, stream>>>(W_ih,  192, pw_ih);
  repack_kernel<<<24, 32, 0, stream>>>(W_hh,  192, pw_hh);

  // 2) Zero scatter accumulators (agg_in and agg_out are contiguous).
  long n4 = (long)2 * N * DIM / 4;
  zero_kernel<<<(int)((n4 + 255) / 256), 256, 0, stream>>>((float4*)agg_in, n4);

  // 3) Linear transforms via WMMA (one wave per 16-row tile).
  int tiles   = (N + 15) / 16;
  int tblocks = (tiles + 7) / 8;
  transform_kernel<<<tblocks, 256, 0, stream>>>(hidden, pw_in, b_in, pw_out, b_out,
                                                t_in, t_out, N);

  // 4) Edge-weighted scatter-sum (L2 atomics), both edge sets.
  long eth = (long)E * 16;
  int eblocks = (int)((eth + 255) / 256);
  scatter_kernel<<<eblocks, 256, 0, stream>>>(t_in,  in_src,  in_dst,  in_w,  agg_in,  E);
  scatter_kernel<<<eblocks, 256, 0, stream>>>(t_out, out_src, out_dst, out_w, agg_out, E);

  // 5) Fused GRU (WMMA gi/gh + gating in registers).
  gru_kernel<<<tblocks, 256, 0, stream>>>(hidden, agg_in, agg_out,
                                          pw_ih, b_ih, pw_hh, b_hh, hy, N);
}
